// LocalPolarPrior_46377056862679
// MI455X (gfx1250) — matrix operand order repaired and checked by reading, not compiled, not run on hardware
//
#include <hip/hip_runtime.h>
#include <hip/hip_bf16.h>
#include <math.h>

typedef __attribute__((ext_vector_type(16))) _Float16 v16h;
typedef __attribute__((ext_vector_type(8)))  _Float16 v8h;
typedef __attribute__((ext_vector_type(8)))  float    v8f;

// Problem constants (fixed by reference setup_inputs)
constexpr int B_  = 8;
constexpr int C_  = 96;    // input channels (K of GEMM)
constexpr int HH  = 384;
constexpr int WW  = 384;
constexpr int PD  = 32;    // projected channels (M of GEMM)
constexpr int WS  = 8;     // window size
constexpr int NR  = 8;     // polar radii
constexpr int NT  = 16;    // polar thetas
constexpr int HG  = HH / WS;   // 48
constexpr int WG  = WW / WS;   // 48
constexpr int COLS = 32;       // columns handled per block (4 windows)
constexpr int NWIN = COLS / WS;    // 4 windows per block
constexpr int NPIX = COLS * WS;    // 256 pixels per block (N of GEMM)
constexpr int KCH  = 32;           // K chunk for WMMA staging

// constexpr twiddles: with fully unrolled loops the (v*x)&7 index is a
// compile-time constant, so these fold to VALU inline constants (0 / ±1 /
// ±sqrt(1/2)) instead of constant-memory loads.
constexpr float K_COS[8] = {
    1.0f, 0.70710678118654752f, 0.0f, -0.70710678118654752f,
   -1.0f, -0.70710678118654752f, 0.0f, 0.70710678118654752f };
constexpr float K_SIN[8] = {
    0.0f, 0.70710678118654752f, 1.0f, 0.70710678118654752f,
    0.0f, -0.70710678118654752f, -1.0f, -0.70710678118654752f };

__launch_bounds__(256, 1)
__global__ void LocalPolarPrior_fused_kernel(const float* __restrict__ x,
                                             const float* __restrict__ Wp,
                                             const float* __restrict__ bp,
                                             float* __restrict__ out)
{
    // ---------------- LDS layout (58.4 KB total) ----------------
    __shared__ __align__(16) _Float16 sA[PD * C_];        // 6144 B  W_proj f16 [m][k]
    __shared__ __align__(32) _Float16 sB[NPIX * KCH];     // 16384 B x chunk f16 [pixel][k]
    __shared__ float sBias[PD];                           // 128 B
    __shared__ float sFeat[PD * NPIX];                    // 32768 B projected windows f32
    __shared__ float sWpol[WS * WS * NT];                 // 4096 B freq-pos -> theta weights
    __shared__ float sLogit[NWIN * NT];                   // 256 B

    const int tid     = threadIdx.x;
    const int wave    = tid >> 5;
    const int lane    = tid & 31;
    const int b       = blockIdx.z;
    const int hg      = blockIdx.y;
    const int cb      = blockIdx.x;            // 0..11 column block
    const int colBase = cb * COLS;
    const int rowBase = hg * WS;

    // ---------------- init: weights to LDS, zero accumulators ----------------
    for (int i = tid; i < PD * C_; i += 256)
        sA[i] = (_Float16)Wp[i];
    if (tid < PD) sBias[tid] = bp[tid];
    for (int i = tid; i < WS * WS * NT; i += 256) sWpol[i] = 0.0f;
    if (tid < NWIN * NT) sLogit[tid] = 0.0f;
    __syncthreads();

    // ---------------- build polar bilinear table (matches _bilinear_weights) --
    if (tid < NR * NT) {
        const int ir = tid / NT;        // radius index (point p = ir*NT + it)
        const int it = tid % NT;        // theta index
        const float radius = (float)ir * (1.0f / (float)(NR - 1));
        const float theta  = 6.283185307179586f * (float)it / (float)NT;
        const float px = (radius * cosf(theta) + 1.0f) * 0.5f * (float)(WS - 1);
        const float py = (radius * sinf(theta) + 1.0f) * 0.5f * (float)(WS - 1);
        const float x0f = floorf(px), y0f = floorf(py);
        const float wx = px - x0f, wy = py - y0f;
        const int xi0 = (int)x0f, yi0 = (int)y0f;
        const int   tx[4] = { xi0, xi0 + 1, xi0,     xi0 + 1 };
        const int   ty[4] = { yi0, yi0,     yi0 + 1, yi0 + 1 };
        const float tw[4] = { (1.f - wx) * (1.f - wy), wx * (1.f - wy),
                              (1.f - wx) * wy,         wx * wy };
        #pragma unroll
        for (int k = 0; k < 4; ++k) {
            const bool valid = (tx[k] >= 0) && (tx[k] < WS) && (ty[k] >= 0) && (ty[k] < WS);
            const int xc = min(max(tx[k], 0), WS - 1);
            const int yc = min(max(ty[k], 0), WS - 1);
            const float wv = valid ? tw[k] : 0.0f;
            atomicAdd(&sWpol[(yc * WS + xc) * NT + it], wv);   // ds_add_f32
        }
    }

    // ---------------- Stage 1: projection GEMM via v_wmma_f32_16x16x32_f16 ----
    // D(32 x 256) = W(32 x 96) * X(96 x 256), K chunked by 32.
    v8f acc[2][2] = {};                       // [mtile][ntile-local], 32 f32 VGPRs
    for (int kc = 0; kc < C_; kc += KCH) {
        // branchless prefetch offset for the next K chunk (re-touches current
        // chunk on the last iteration, which is harmless)
        const size_t pfOff = (kc + KCH < C_) ? (size_t)KCH * HH * WW : 0;

        // stage x chunk -> sB as f16, [pixel(=col*8+row)][k].
        // float4 (b128) loads: 2048 float4 units, 8 per thread, coalesced.
        #pragma unroll
        for (int it = 0; it < (KCH * NPIX / 4) / 256; ++it) {
            const int i   = tid + it * 256;
            const int c   = i >> 6;           // 0..31 channel within chunk
            const int rem = i & 63;
            const int yy  = rem >> 3;         // 0..7 row within window-row
            const int xc4 = rem & 7;          // 0..7 group of 4 columns
            const float* gp = x + (((size_t)(b * C_ + kc + c) * HH) + rowBase + yy) * WW
                                + colBase + xc4 * 4;
            const float4 g = *(const float4*)gp;          // global_load_b128
            __builtin_prefetch(gp + pfOff, 0, 1);         // global_prefetch_b8
            const int p0 = (xc4 * 4) * WS + yy;           // pixel = col*8 + row
            sB[(p0 + 0 * WS) * KCH + c] = (_Float16)g.x;
            sB[(p0 + 1 * WS) * KCH + c] = (_Float16)g.y;
            sB[(p0 + 2 * WS) * KCH + c] = (_Float16)g.z;
            sB[(p0 + 3 * WS) * KCH + c] = (_Float16)g.w;
        }
        __syncthreads();

        #pragma unroll
        for (int mt = 0; mt < 2; ++mt) {
            // A tile 16x32 f16: lane row = mt*16 + (lane&15); 16 contiguous K per half-wave
            const int arow = mt * 16 + (lane & 15);
            const int kb   = (lane >> 4) * 16;
            const v8h lo = *(const v8h*)&sA[arow * C_ + kc + kb];
            const v8h hi = *(const v8h*)&sA[arow * C_ + kc + kb + 8];
            const v16h a = __builtin_shufflevector(lo, hi,
                0, 1, 2, 3, 4, 5, 6, 7, 8, 9, 10, 11, 12, 13, 14, 15);
            #pragma unroll
            for (int nt2 = 0; nt2 < 2; ++nt2) {
                // B tile 32x16 f16: lane col = ntile*16 + (lane&15); 16 contiguous K
                const int ntile = wave * 2 + nt2;
                const int pix   = ntile * 16 + (lane & 15);
                const int ks    = (lane >> 4) * 16;
                const v16h bm = *(const v16h*)&sB[pix * KCH + ks];
                acc[mt][nt2] = __builtin_amdgcn_wmma_f32_16x16x32_f16(
                    false, a, false, bm, (short)0, acc[mt][nt2], false, false);
            }
        }
        __syncthreads();
    }

    // scatter D (+bias) into sFeat[channel][pixel]
    #pragma unroll
    for (int mt = 0; mt < 2; ++mt) {
        #pragma unroll
        for (int nt2 = 0; nt2 < 2; ++nt2) {
            const int ntile = wave * 2 + nt2;
            const int npix  = ntile * 16 + (lane & 15);
            #pragma unroll
            for (int r = 0; r < 8; ++r) {
                const int m = mt * 16 + (lane >> 4) * 8 + r;
                sFeat[m * NPIX + npix] = acc[mt][nt2][r] + sBias[m];
            }
        }
    }
    __syncthreads();

    // ---------------- Stage 2: 8x8 FFT + log1p|.| + polar -> theta sums ------
    if (tid < NWIN * PD) {
        const int w  = tid >> 5;              // window 0..3
        const int ch = tid & 31;              // channel 0..31
        const float* f = &sFeat[ch * NPIX + w * 64];  // [x*8 + y]

        float Rr[8][8], Ri[8][8];             // row DFT result, [y][v]
        #pragma unroll
        for (int y = 0; y < 8; ++y) {
            float row[8];
            #pragma unroll
            for (int xx = 0; xx < 8; ++xx) row[xx] = f[xx * 8 + y];
            #pragma unroll
            for (int v = 0; v < 8; ++v) {
                float cr = 0.f, ci = 0.f;
                #pragma unroll
                for (int xx = 0; xx < 8; ++xx) {
                    const int k = (v * xx) & 7;
                    cr += row[xx] * K_COS[k];
                    ci -= row[xx] * K_SIN[k];
                }
                Rr[y][v] = cr; Ri[y][v] = ci;
            }
        }

        float st[NT];
        #pragma unroll
        for (int t = 0; t < NT; ++t) st[t] = 0.f;

        #pragma unroll
        for (int u = 0; u < 8; ++u) {
            #pragma unroll
            for (int v = 0; v < 8; ++v) {
                float sr = 0.f, si = 0.f;
                #pragma unroll
                for (int y = 0; y < 8; ++y) {
                    const int k = (u * y) & 7;
                    sr += Rr[y][v] * K_COS[k] + Ri[y][v] * K_SIN[k];  // e^{-i2pi uy/8} * R
                    si += Ri[y][v] * K_COS[k] - Rr[y][v] * K_SIN[k];
                }
                // ortho norm 1/8; fftshift folded into position index
                const float mag = log1pf(0.125f * sqrtf(sr * sr + si * si));
                const int pos = (((u + 4) & 7) * 8 + ((v + 4) & 7));
                const float* wp = &sWpol[pos * NT];
                #pragma unroll
                for (int t = 0; t < NT; ++t) st[t] += mag * wp[t];
            }
        }
        #pragma unroll
        for (int t = 0; t < NT; ++t)
            atomicAdd(&sLogit[w * NT + t], st[t]);       // ds_add_f32
    }
    __syncthreads();

    // ---------------- Stage 3: softmax / entropy / confidence -----------------
    if (tid < NWIN) {
        const int w    = tid;
        const int wcol = cb * NWIN + w;
        float lg[NT];
        float mx = -1e30f;
        #pragma unroll
        for (int t = 0; t < NT; ++t) {
            lg[t] = sLogit[w * NT + t] * (1.0f / (float)(PD * NR));  // mean over (pd, r)
            mx = fmaxf(mx, lg[t]);
        }
        float sum = 0.f;
        #pragma unroll
        for (int t = 0; t < NT; ++t) { lg[t] = expf(lg[t] - mx); sum += lg[t]; }
        const float inv = 1.0f / sum;
        float ent = 0.f;
        #pragma unroll
        for (int t = 0; t < NT; ++t) {
            const float d = lg[t] * inv;
            out[(((size_t)b * NT + t) * HG + hg) * WG + wcol] = d;   // d_local
            ent -= d * logf(fmaxf(d, 1e-8f));
        }
        float conf = 1.0f - ent * (1.0f / 2.772588722239781f);       // / log(16)
        conf = fminf(fmaxf(conf, 0.0f), 1.0f);
        out[(size_t)B_ * NT * HG * WG + ((size_t)b * HG + hg) * WG + wcol] = conf;
    }
}

extern "C" void kernel_launch(void* const* d_in, const int* in_sizes, int n_in,
                              void* d_out, int out_size, void* d_ws, size_t ws_size,
                              hipStream_t stream) {
    const float* x  = (const float*)d_in[0];   // (8, 96, 384, 384) f32
    const float* Wp = (const float*)d_in[1];   // (32, 96) f32
    const float* bp = (const float*)d_in[2];   // (32,) f32
    float* out = (float*)d_out;                // d_local (294912) ++ confidence (18432)

    dim3 grid(WW / COLS, HG, B_);              // (12, 48, 8) = 4608 blocks
    dim3 block(256);                           // 8 waves of 32
    hipLaunchKernelGGL(LocalPolarPrior_fused_kernel, grid, block, 0, stream,
                       x, Wp, bp, out);
}
